// Label_Relation_68006512165282
// MI455X (gfx1250) — matrix operand order repaired and checked by reading, compiled, not run on hardware
//
#include <hip/hip_runtime.h>

// ---------------------------------------------------------------------------
// Problem constants (from reference): B=8, C=32, H=W=256, NC=5, HW=65536
// ---------------------------------------------------------------------------
#define HW_   65536
#define B_    8
#define C_    32
#define NC_   5

typedef __attribute__((ext_vector_type(16))) __bf16        v16bf;
typedef __attribute__((ext_vector_type(8)))  float         v8f;
typedef __attribute__((ext_vector_type(8)))  unsigned int  u32x8;

__device__ __forceinline__ unsigned short f2bf(float f) {
    union { float f; unsigned u; } v; v.f = f;
    unsigned r = v.u + 0x7FFFu + ((v.u >> 16) & 1u);   // round-to-nearest-even
    return (unsigned short)(r >> 16);
}

// LDS offset of a __shared__ pointer (flat LDS-aperture addr keeps the group
// offset in the low 32 bits; async-to-LDS VDST wants the LDS byte address).
__device__ __forceinline__ unsigned lds_off_u32(const void* p) {
    return (unsigned)(unsigned long long)p;
}

// global_load_async_to_lds_b128: copies 16B/lane memory -> LDS, ASYNCcnt.
__device__ __forceinline__ void async_ld_b128(unsigned lds_addr, unsigned goff,
                                              const void* sbase) {
    asm volatile("global_load_async_to_lds_b128 %0, %1, %2"
                 :
                 : "v"(lds_addr), "v"(goff), "s"(sbase)
                 : "memory");
}
__device__ __forceinline__ void wait_asynccnt0() {
    asm volatile("s_wait_asynccnt 0x0" ::: "memory");
}

// ---------------------------------------------------------------------------
// Shared WMMA mainloop: A patch in sA [5][68][32] bf16, weights sW [25][OC][32]
// bf16. One wave computes 16 pixels x 16*OCT out-channels.
// ---------------------------------------------------------------------------
template<int OCT>
__device__ __forceinline__ void conv_mainloop(
    const unsigned short* sA, const unsigned short* sW,
    int mbase, int l15, int hi, v8f (&acc)[OCT])
{
    constexpr int OC = OCT * 16;
    #pragma unroll
    for (int tap = 0; tap < 25; ++tap) {
        const int ky = tap / 5, kx = tap % 5;
        // A: lane<16 holds K {0..7,16..23}; lane>=16 holds K {8..15,24..31}
        const unsigned short* ap =
            &sA[(ky * 68 + (mbase + l15 + kx)) * 32 + (hi ? 8 : 0)];
        uint4 alo = *(const uint4*)(ap);
        uint4 ahi = *(const uint4*)(ap + 16);
        u32x8 au;
        au[0]=alo.x; au[1]=alo.y; au[2]=alo.z; au[3]=alo.w;
        au[4]=ahi.x; au[5]=ahi.y; au[6]=ahi.z; au[7]=ahi.w;
        v16bf A = __builtin_bit_cast(v16bf, au);

        #pragma unroll
        for (int nt = 0; nt < OCT; ++nt) {
            // B: lane<16 holds K {0..7,16..23}; lane>=16 holds K {8..15,24..31}
            const unsigned short* bp =
                &sW[(tap * OC + (nt * 16 + l15)) * 32 + (hi ? 8 : 0)];
            uint4 blo = *(const uint4*)(bp);
            uint4 bhi = *(const uint4*)(bp + 16);
            u32x8 bu;
            bu[0]=blo.x; bu[1]=blo.y; bu[2]=blo.z; bu[3]=blo.w;
            bu[4]=bhi.x; bu[5]=bhi.y; bu[6]=bhi.z; bu[7]=bhi.w;
            v16bf Bv = __builtin_bit_cast(v16bf, bu);

            acc[nt] = __builtin_amdgcn_wmma_f32_16x16x32_bf16(
                false, A, false, Bv, (short)0, acc[nt], false, false);
        }
    }
}

// ---------------------------------------------------------------------------
// Proj conv (f32 NCHW input): 5x5 SAME conv 32->32, writes f32 NCHW output and
// (optionally) a bf16 NHWC shadow copy for later async-staged convs.
// ---------------------------------------------------------------------------
template<int OCT, int OCS>
__global__ __launch_bounds__(128) void conv5x5_wmma_f32in(
    const float* __restrict__ x, const float* __restrict__ w,
    const float* __restrict__ bias, float* __restrict__ y,
    unsigned short* __restrict__ ybf)
{
    constexpr int OC = OCT * 16;
    __shared__ __align__(16) unsigned short sA[5 * 68 * 32];   // [row][col][ic]
    __shared__ __align__(16) unsigned short sW[25 * OC * 32];  // [tap][oc][ic]

    const int tid = threadIdx.x;
    const int bx  = blockIdx.x;   // 64-pixel strip along W
    const int by  = blockIdx.y;   // H row
    const int bz  = blockIdx.z;   // batch

    for (int i = tid; i < 5 * 68 * 32; i += 128) {
        int c   = i / (5 * 68);
        int rem = i % (5 * 68);
        int r   = rem / 68;
        int col = rem % 68;
        int yy = by + r - 2;
        int xg = bx * 64 + col - 2;
        float v = 0.0f;
        if ((unsigned)yy < 256u && (unsigned)xg < 256u)
            v = x[((bz * 32 + c) * 256 + yy) * 256 + xg];
        sA[(r * 68 + col) * 32 + c] = f2bf(v);
    }
    for (int i = tid; i < 25 * OC * 32; i += 128) {
        int oc  = i / 800;
        int rem = i % 800;
        int ic  = rem / 25;
        int tap = rem % 25;
        float v = (oc < OCS) ? w[i] : 0.0f;
        sW[(tap * OC + oc) * 32 + ic] = f2bf(v);
    }
    __syncthreads();

    const int lane  = tid & 31;
    const int wave  = tid >> 5;
    const int mbase = wave * 16;
    const int hi    = lane >> 4;
    const int l15   = lane & 15;

    v8f acc[OCT] = {};
    conv_mainloop<OCT>(sA, sW, mbase, l15, hi, acc);

    const int mrow = hi * 8;
    #pragma unroll
    for (int nt = 0; nt < OCT; ++nt) {
        int oc = nt * 16 + l15;
        if (oc < OCS) {
            float bv = bias[oc];
            float* dst = y + ((bz * OCS + oc) * 256 + by) * 256
                           + bx * 64 + mbase + mrow;
            float4 o0 = make_float4(acc[nt][0]+bv, acc[nt][1]+bv, acc[nt][2]+bv, acc[nt][3]+bv);
            float4 o1 = make_float4(acc[nt][4]+bv, acc[nt][5]+bv, acc[nt][6]+bv, acc[nt][7]+bv);
            *(float4*)(dst)     = o0;
            *(float4*)(dst + 4) = o1;
            if (ybf) {   // bf16 NHWC shadow for async-staged consumers
                unsigned long long base =
                    (((unsigned long long)bz * 65536ull
                      + (unsigned long long)(by * 256 + bx * 64 + mbase + mrow)) * 32ull)
                    + (unsigned)oc;
                #pragma unroll
                for (int v = 0; v < 8; ++v)
                    ybf[base + (unsigned long long)v * 32ull] = f2bf(acc[nt][v] + bv);
            }
        }
    }
}

// ---------------------------------------------------------------------------
// Head conv (bf16 NHWC input): activations staged straight into LDS with
// GLOBAL_LOAD_ASYNC_TO_LDS_B128 (halo pre-zeroed; OOB lanes masked off).
// ---------------------------------------------------------------------------
template<int OCT, int OCS>
__global__ __launch_bounds__(128) void conv5x5_wmma_async(
    const unsigned short* __restrict__ xbf, const float* __restrict__ w,
    const float* __restrict__ bias, float* __restrict__ y)
{
    constexpr int OC = OCT * 16;
    __shared__ __align__(16) unsigned short sA[5 * 68 * 32];   // [row][col][ic]
    __shared__ __align__(16) unsigned short sW[25 * OC * 32];  // [tap][oc][ic]

    const int tid = threadIdx.x;
    const int bx  = blockIdx.x;
    const int by  = blockIdx.y;
    const int bz  = blockIdx.z;

    // 1) zero the whole patch (provides SAME padding), 16B per chunk
    const uint4 z4 = make_uint4(0u, 0u, 0u, 0u);
    for (int i = tid; i < 1360; i += 128)          // 1360 * 16B = 5*68*32*2B
        ((uint4*)sA)[i] = z4;
    __syncthreads();

    // 2) async-copy in-bounds 16B chunks (8 channels) memory -> LDS
    const unsigned sAbase = lds_off_u32(sA);
    const unsigned short* src = xbf + (unsigned long long)bz * 65536ull * 32ull;
    for (int i = tid; i < 1360; i += 128) {
        int r   = i / 272;          // 272 = 68 cols * 4 chunks
        int t   = i % 272;
        int col = t >> 2;
        int q   = t & 3;            // which group of 8 channels
        int yy = by + r - 2;
        int xg = bx * 64 + col - 2;
        if ((unsigned)yy < 256u && (unsigned)xg < 256u) {
            unsigned lds  = sAbase + (unsigned)((r * 68 + col) * 64 + q * 16);
            unsigned goff = (unsigned)(((yy * 256 + xg) * 32 + q * 8) * 2);
            async_ld_b128(lds, goff, src);
        }
    }

    // 3) stage weights (small) while async copies are in flight
    for (int i = tid; i < 25 * OC * 32; i += 128) {
        int oc  = i / 800;
        int rem = i % 800;
        int ic  = rem / 25;
        int tap = rem % 25;
        float v = (oc < OCS) ? w[i] : 0.0f;
        sW[(tap * OC + oc) * 32 + ic] = f2bf(v);
    }
    wait_asynccnt0();
    __syncthreads();

    const int lane  = tid & 31;
    const int wave  = tid >> 5;
    const int mbase = wave * 16;
    const int hi    = lane >> 4;
    const int l15   = lane & 15;

    v8f acc[OCT] = {};
    conv_mainloop<OCT>(sA, sW, mbase, l15, hi, acc);

    const int mrow = hi * 8;
    #pragma unroll
    for (int nt = 0; nt < OCT; ++nt) {
        int oc = nt * 16 + l15;
        if (oc < OCS) {
            float bv = bias[oc];
            float* dst = y + ((bz * OCS + oc) * 256 + by) * 256
                           + bx * 64 + mbase + mrow;
            float4 o0 = make_float4(acc[nt][0]+bv, acc[nt][1]+bv, acc[nt][2]+bv, acc[nt][3]+bv);
            float4 o1 = make_float4(acc[nt][4]+bv, acc[nt][5]+bv, acc[nt][6]+bv, acc[nt][7]+bv);
            *(float4*)(dst)     = o0;
            *(float4*)(dst + 4) = o1;
        }
    }
}

// ---------------------------------------------------------------------------
// fea[b,l,c] = sum_p fs'[b,c,p] * label[b,l,p] * w_lin[p]   (then /B + b_lin)
// ---------------------------------------------------------------------------
__global__ void zero_small(float* p, int n) {
    int i = blockIdx.x * blockDim.x + threadIdx.x;
    if (i < n) p[i] = 0.0f;
}

__global__ __launch_bounds__(160) void fea_accum(
    const float* __restrict__ fsp, const float* __restrict__ lab,
    const float* __restrict__ wlin, float* __restrict__ fsum)
{
    const int l = threadIdx.x / 32, c = threadIdx.x % 32;
    const int b = blockIdx.x;
    const int p0 = blockIdx.y * 2048;
    const float* fr = fsp + (b * 32 + c) * HW_;
    const float* lr = lab + (b * 5 + l) * HW_;
    float acc = 0.0f;
    for (int p = p0; p < p0 + 2048; ++p)
        acc += fr[p] * lr[p] * wlin[p];
    atomicAdd(&fsum[l * 32 + c], acc);
}

__global__ void fea_final(const float* __restrict__ fsum,
                          const float* __restrict__ blin,
                          float* __restrict__ fmf)
{
    int i = threadIdx.x;
    if (i < 160) fmf[i] = fsum[i] * (1.0f / B_) + blin[0];
}

// ---------------------------------------------------------------------------
// Per-pixel attention; emits bf16 NHWC directly (consumed by async conv).
//   s[l] = 32 * <q, fea_mf[l]>; softmax over l; out[c] = sum_l w[l]*F[l][c]+q[c]
// ---------------------------------------------------------------------------
__global__ __launch_bounds__(256) void attn(
    const float* __restrict__ fqp, const float* __restrict__ fmf,
    unsigned short* __restrict__ out_bf)
{
    __shared__ float sF[160];
    if (threadIdx.x < 160) sF[threadIdx.x] = fmf[threadIdx.x];
    __syncthreads();

    const int g = blockIdx.x * 256 + threadIdx.x;
    const int b = g >> 16;
    const int p = g & 65535;

    float q[32];
    #pragma unroll
    for (int c = 0; c < 32; ++c) q[c] = fqp[(b * 32 + c) * HW_ + p];

    float s[5];
    #pragma unroll
    for (int l = 0; l < 5; ++l) {
        float d = 0.0f;
        #pragma unroll
        for (int c = 0; c < 32; ++c) d += q[c] * sF[l * 32 + c];
        s[l] = d * 32.0f;             // scale = C (faithful to reference bug)
    }
    float m = s[0];
    #pragma unroll
    for (int l = 1; l < 5; ++l) m = fmaxf(m, s[l]);
    float sum = 0.0f;
    #pragma unroll
    for (int l = 0; l < 5; ++l) { s[l] = __expf(s[l] - m); sum += s[l]; }
    const float inv = 1.0f / sum;

    unsigned pk[16];
    #pragma unroll
    for (int c2 = 0; c2 < 16; ++c2) {
        float o0 = 0.0f, o1 = 0.0f;
        #pragma unroll
        for (int l = 0; l < 5; ++l) {
            o0 += s[l] * sF[l * 32 + 2 * c2];
            o1 += s[l] * sF[l * 32 + 2 * c2 + 1];
        }
        o0 = o0 * inv + q[2 * c2];
        o1 = o1 * inv + q[2 * c2 + 1];
        pk[c2] = (unsigned)f2bf(o0) | ((unsigned)f2bf(o1) << 16);
    }
    uint4* dst = (uint4*)(out_bf + (unsigned long long)g * 32ull);
    dst[0] = make_uint4(pk[0],  pk[1],  pk[2],  pk[3]);
    dst[1] = make_uint4(pk[4],  pk[5],  pk[6],  pk[7]);
    dst[2] = make_uint4(pk[8],  pk[9],  pk[10], pk[11]);
    dst[3] = make_uint4(pk[12], pk[13], pk[14], pk[15]);
}

// ---------------------------------------------------------------------------
extern "C" void kernel_launch(void* const* d_in, const int* in_sizes, int n_in,
                              void* d_out, int out_size, void* d_ws, size_t ws_size,
                              hipStream_t stream)
{
    (void)in_sizes; (void)n_in; (void)out_size; (void)ws_size;

    const float* fs    = (const float*)d_in[0];
    const float* fq    = (const float*)d_in[1];
    const float* lab   = (const float*)d_in[2];
    const float* w_ps  = (const float*)d_in[3];
    const float* b_ps  = (const float*)d_in[4];
    const float* w_pq  = (const float*)d_in[5];
    const float* b_pq  = (const float*)d_in[6];
    const float* w_lin = (const float*)d_in[7];
    const float* b_lin = (const float*)d_in[8];
    const float* w_cs  = (const float*)d_in[9];
    const float* b_cs  = (const float*)d_in[10];
    const float* w_cq  = (const float*)d_in[11];
    const float* b_cq  = (const float*)d_in[12];

    const size_t TENF  = (size_t)B_ * C_ * HW_ * sizeof(float);          // 64 MB
    const size_t TENBF = (size_t)B_ * C_ * HW_ * sizeof(unsigned short); // 32 MB
    char* ws = (char*)d_ws;
    float*          fs_p  = (float*)(ws);
    float*          fq_p  = (float*)(ws + TENF);
    unsigned short* fs_pb = (unsigned short*)(ws + 2 * TENF);
    unsigned short* lgt_b = (unsigned short*)(ws + 2 * TENF + TENBF);
    float*          fsum  = (float*)(ws + 2 * TENF + 2 * TENBF);
    float*          fmf   = fsum + 256;

    float* out_s = (float*)d_out;
    float* out_q = out_s + (size_t)B_ * NC_ * HW_;

    dim3 cg(4, 256, 8);   // W strips x H rows x batch
    conv5x5_wmma_f32in<2, 32><<<cg, 128, 0, stream>>>(fs, w_ps, b_ps, fs_p, fs_pb);
    conv5x5_wmma_f32in<2, 32><<<cg, 128, 0, stream>>>(fq, w_pq, b_pq, fq_p, nullptr);

    zero_small<<<1, 192, 0, stream>>>(fsum, 160);
    fea_accum<<<dim3(8, 32), 160, 0, stream>>>(fs_p, lab, w_lin, fsum);
    fea_final<<<1, 160, 0, stream>>>(fsum, b_lin, fmf);

    attn<<<2048, 256, 0, stream>>>(fq_p, fmf, lgt_b);

    conv5x5_wmma_async<1, 5><<<cg, 128, 0, stream>>>(fs_pb, w_cs, b_cs, out_s);
    conv5x5_wmma_async<1, 5><<<cg, 128, 0, stream>>>(lgt_b, w_cq, b_cq, out_q);
}
